// TemporalAttentionLayer_45612552683795
// MI455X (gfx1250) — matrix operand order repaired and checked by reading, not compiled, not run on hardware
//
#include <hip/hip_runtime.h>

#define ALPHA 0.2f

typedef __attribute__((ext_vector_type(16))) __bf16 v16bf;
typedef __attribute__((ext_vector_type(8)))  __bf16 v8bf;
typedef __attribute__((ext_vector_type(8)))  float  v8f;

static __device__ __forceinline__ v16bf concat8(v8bf lo, v8bf hi) {
  return __builtin_shufflevector(lo, hi, 0,1,2,3,4,5,6,7,8,9,10,11,12,13,14,15);
}

// ---------------- Kernel 1: s1 = x@w1, s2 = x@w2 (one wave per row) ----------
__global__ __launch_bounds__(256) void gat_s_kernel(
    const float* __restrict__ x, const float* __restrict__ w,
    float* __restrict__ s1, float* __restrict__ s2, int N, int K)
{
  const int wave = threadIdx.x >> 5;
  const int lane = threadIdx.x & 31;
  const int row  = blockIdx.x * 8 + wave;
  if (row >= K) return;
  const float* xr = x + (size_t)row * N;
  float d1 = 0.f, d2 = 0.f;
  for (int k = lane; k < N; k += 32) {
    float xv = xr[k];
    d1 += xv * w[k];
    d2 += xv * w[N + k];
  }
  #pragma unroll
  for (int off = 16; off >= 1; off >>= 1) {
    d1 += __shfl_xor(d1, off, 32);
    d2 += __shfl_xor(d2, off, 32);
  }
  if (lane == 0) { s1[row] = d1; s2[row] = d2; }
}

// ---------------- Kernel 2: s2max = max(s2) ---------------------------------
__global__ __launch_bounds__(256) void gat_max_kernel(
    const float* __restrict__ s2, float* __restrict__ s2max, int K)
{
  __shared__ float red[256];
  float m = -3.4e38f;
  for (int i = threadIdx.x; i < K; i += 256) m = fmaxf(m, s2[i]);
  red[threadIdx.x] = m;
  __syncthreads();
  #pragma unroll
  for (int s = 128; s > 0; s >>= 1) {
    if (threadIdx.x < s) red[threadIdx.x] = fmaxf(red[threadIdx.x], red[threadIdx.x + s]);
    __syncthreads();
  }
  if (threadIdx.x == 0) s2max[0] = red[0];
}

// ---------------- Kernel 2b: xT[n][j] = bf16(x[j][n]) -----------------------
__global__ __launch_bounds__(256) void gat_tr_kernel(
    const float* __restrict__ x, void* __restrict__ xTv, int N, int K)
{
  __bf16* xT = (__bf16*)xTv;
  __shared__ float tile[32][33];
  const int tx = threadIdx.x & 31;
  const int ty = threadIdx.x >> 5;            // 0..7
  const int j0 = blockIdx.x * 32;
  const int n0 = blockIdx.y * 32;
  #pragma unroll
  for (int i = 0; i < 4; ++i)
    tile[ty + 8 * i][tx] = x[(size_t)(j0 + ty + 8 * i) * N + n0 + tx];
  __syncthreads();
  #pragma unroll
  for (int i = 0; i < 4; ++i)
    xT[(size_t)(n0 + ty + 8 * i) * K + j0 + tx] = (__bf16)tile[tx][ty + 8 * i];
}

// ---------------- Kernel 3: fused softmax(e) @ x via bf16 WMMA --------------
// Block: 256 threads (8 waves). Tile: 32 rows x 256 cols of h. j-step 64.
// Ping-pong P tile in LDS -> one barrier per iteration.
template<bool USE_XT>
__global__ __launch_bounds__(256) void gat_attn_kernel(
    const float* __restrict__ x, const void* __restrict__ xTv,
    const float* __restrict__ s1g, const float* __restrict__ s2g,
    const float* __restrict__ s2maxp, float* __restrict__ h, int N, int K)
{
  constexpr int R = 32, JSTEP = 64, PSTR = 72; // 144B row stride: 16B-aligned, conflict-free
  __shared__ __align__(16) __bf16 Pbuf[2][R * PSTR];
  __shared__ float Zfin[R];
  const __bf16* xT = (const __bf16*)xTv;

  const int tid = threadIdx.x, lane = tid & 31, wave = tid >> 5;
  const int r0 = blockIdx.y * R, n0 = blockIdx.x * 256;

  // --- P-producer role: row pr (0..31), 8 consecutive j at offset pj --------
  const int pr = tid >> 3;
  const int pj = (tid & 7) * 8;
  const float s2max = s2maxp[0];
  const float s1r = s1g[r0 + pr];
  const float tmx = s1r + s2max;
  const float mr  = tmx > 0.f ? tmx : ALPHA * tmx;   // exact row max (lrelu monotone)
  float zpriv = 0.f;

  // p = exp(lrelu(s1+s2) - m); writes one 16B ds_store_b128
  auto produce = [&](int jbase, __bf16* dst) {
    const float4 sa = *(const float4*)(s2g + jbase + pj);
    const float4 sb = *(const float4*)(s2g + jbase + pj + 4);
    const float ev[8] = {sa.x, sa.y, sa.z, sa.w, sb.x, sb.y, sb.z, sb.w};
    v8bf pv;
    #pragma unroll
    for (int q = 0; q < 8; ++q) {
      float t = s1r + ev[q];
      t = t > 0.f ? t : ALPHA * t;
      float p = __expf(t - mr);
      zpriv += p;
      pv[q] = (__bf16)p;
    }
    *(v8bf*)(dst + pr * PSTR + pj) = pv;
  };

  // --- WMMA fragment roles --------------------------------------------------
  const int halfSel = (lane < 16) ? 0 : 1;
  const int mrow = lane & 15;
  const int kA = halfSel * 16;      // 16-bit A 16x32 per-lane K offset
  const int kB = halfSel * 16;      // 16-bit B 32x16 per-lane K base
  const int cbase = n0 + wave * 32 + mrow;

  const __bf16* xr0 = USE_XT ? xT + (size_t)cbase * K + kB : nullptr;
  const __bf16* xr1 = USE_XT ? xT + (size_t)(cbase + 16) * K + kB : nullptr;

  v8f acc00{}, acc01{}, acc10{}, acc11{};

  // prologue: tile 0 into buffer 0
  produce(0, &Pbuf[0][0]);
  __syncthreads();

  for (int j0 = 0; j0 < K; j0 += JSTEP) {
    const int cur = (j0 / JSTEP) & 1;
    const __bf16* Pc = &Pbuf[cur][0];

    // ---- produce NEXT tile into the alternate buffer (no barrier needed) ---
    if (j0 + JSTEP < K) produce(j0 + JSTEP, &Pbuf[cur ^ 1][0]);

    // ---- A fragments from LDS (8x ds_load_b128) ----------------------------
    v16bf A00, A01, A10, A11;                // [row-tile][k-block]
    {
      const __bf16* p0 = Pc + mrow * PSTR + kA;
      const __bf16* p1 = Pc + (16 + mrow) * PSTR + kA;
      A00 = concat8(*(const v8bf*)(p0),      *(const v8bf*)(p0 + 8));
      A01 = concat8(*(const v8bf*)(p0 + 32), *(const v8bf*)(p0 + 40));
      A10 = concat8(*(const v8bf*)(p1),      *(const v8bf*)(p1 + 8));
      A11 = concat8(*(const v8bf*)(p1 + 32), *(const v8bf*)(p1 + 40));
    }

    // ---- B fragments --------------------------------------------------------
    v16bf B00, B01, B10, B11;                // [col-tile][k-block]
    if constexpr (USE_XT) {
      B00 = concat8(*(const v8bf*)(xr0 + j0),      *(const v8bf*)(xr0 + j0 + 8));
      B01 = concat8(*(const v8bf*)(xr0 + j0 + 32), *(const v8bf*)(xr0 + j0 + 40));
      B10 = concat8(*(const v8bf*)(xr1 + j0),      *(const v8bf*)(xr1 + j0 + 8));
      B11 = concat8(*(const v8bf*)(xr1 + j0 + 32), *(const v8bf*)(xr1 + j0 + 40));
      __builtin_prefetch(xr0 + j0 + JSTEP, 0, 3);  // WGP-scope global_prefetch_b8
      __builtin_prefetch(xr1 + j0 + JSTEP, 0, 3);
    } else {
      const float* xb0 = x + (size_t)(j0 + kB) * N;
      const float* xb1 = x + (size_t)(j0 + 32 + kB) * N;
      #pragma unroll
      for (int v = 0; v < 16; ++v) {
        B00[v] = (__bf16)xb0[(size_t)v * N + cbase];
        B01[v] = (__bf16)xb1[(size_t)v * N + cbase];
        B10[v] = (__bf16)xb0[(size_t)v * N + cbase + 16];
        B11[v] = (__bf16)xb1[(size_t)v * N + cbase + 16];
      }
    }

    // ---- 8 WMMAs ------------------------------------------------------------
    acc00 = __builtin_amdgcn_wmma_f32_16x16x32_bf16(false, A00, false, B00, (short)0, acc00, false, false);
    acc00 = __builtin_amdgcn_wmma_f32_16x16x32_bf16(false, A01, false, B01, (short)0, acc00, false, false);
    acc01 = __builtin_amdgcn_wmma_f32_16x16x32_bf16(false, A00, false, B10, (short)0, acc01, false, false);
    acc01 = __builtin_amdgcn_wmma_f32_16x16x32_bf16(false, A01, false, B11, (short)0, acc01, false, false);
    acc10 = __builtin_amdgcn_wmma_f32_16x16x32_bf16(false, A10, false, B00, (short)0, acc10, false, false);
    acc10 = __builtin_amdgcn_wmma_f32_16x16x32_bf16(false, A11, false, B01, (short)0, acc10, false, false);
    acc11 = __builtin_amdgcn_wmma_f32_16x16x32_bf16(false, A10, false, B10, (short)0, acc11, false, false);
    acc11 = __builtin_amdgcn_wmma_f32_16x16x32_bf16(false, A11, false, B11, (short)0, acc11, false, false);

    // one barrier per iteration: orders P(next) writes vs next-iter reads AND
    // this-iter A reads vs next-iter overwrite
    __syncthreads();
  }

  // ---- finalize Z per row (8 producer threads per row) ----------------------
  zpriv += __shfl_xor(zpriv, 1, 32);
  zpriv += __shfl_xor(zpriv, 2, 32);
  zpriv += __shfl_xor(zpriv, 4, 32);
  if ((tid & 7) == 0) Zfin[pr] = zpriv;
  __syncthreads();

  // ---- epilogue: scale by 1/Z and store (C/D 16x16 f32 layout) --------------
  #pragma unroll
  for (int g = 0; g < 8; ++g) {
    const int rl0 = g + halfSel * 8;
    const float rz0 = 1.0f / Zfin[rl0];
    h[(size_t)(r0 + rl0) * N + cbase]      = acc00[g] * rz0;
    h[(size_t)(r0 + rl0) * N + cbase + 16] = acc01[g] * rz0;
    const int rl1 = 16 + rl0;
    const float rz1 = 1.0f / Zfin[rl1];
    h[(size_t)(r0 + rl1) * N + cbase]      = acc10[g] * rz1;
    h[(size_t)(r0 + rl1) * N + cbase + 16] = acc11[g] * rz1;
  }
}

extern "C" void kernel_launch(void* const* d_in, const int* in_sizes, int n_in,
                              void* d_out, int out_size, void* d_ws, size_t ws_size,
                              hipStream_t stream) {
  const float* x = (const float*)d_in[0];
  const float* w = (const float*)d_in[1];
  const int N = in_sizes[1] / 2;        // 1024
  const int K = in_sizes[0] / N;        // 4096
  float* s1  = (float*)d_ws;
  float* s2  = s1 + K;
  float* s2m = s2 + K;
  float* h   = (float*)d_out;

  const size_t scalarBytes = (((size_t)(2 * K + 1) * sizeof(float)) + 255) & ~(size_t)255;
  void* xT = (void*)((char*)d_ws + scalarBytes);
  const size_t need = scalarBytes + (size_t)N * K * 2;

  gat_s_kernel<<<dim3((K + 7) / 8), 256, 0, stream>>>(x, w, s1, s2, N, K);
  gat_max_kernel<<<dim3(1), 256, 0, stream>>>(s2, s2m, K);
  if (ws_size >= need) {
    gat_tr_kernel<<<dim3(K / 32, N / 32), 256, 0, stream>>>(x, xT, N, K);
    gat_attn_kernel<true><<<dim3(N / 256, K / 32), 256, 0, stream>>>(x, xT, s1, s2, s2m, h, N, K);
  } else {
    gat_attn_kernel<false><<<dim3(N / 256, K / 32), 256, 0, stream>>>(x, xT, s1, s2, s2m, h, N, K);
  }
}
